// Attention2D_45715631899465
// MI455X (gfx1250) — compile-verified, matrix operand order checked by reading, not run on hardware
//
#include <hip/hip_runtime.h>

// MI455X / gfx1250 fused kernel for the reference "Attention2D" module.
// Fully fused: one pass over k (134MB) -> memory-bound near the 23.3TB/s roofline.
// All matmuls via v_wmma_f32_16x16x32_bf16 (wave32 WMMA), softmax-over-V done
// entirely in-register per lane (V=8 maps onto the 8 C-fragment VGPRs).

#define DIMM 64
#define BN_TOTAL 65536   // B*N = 1024*64
#define KROWS 524288     // B*N*V

typedef __attribute__((ext_vector_type(16))) __bf16 v16bf;
typedef __attribute__((ext_vector_type(8)))  __bf16 v8bf;
typedef __attribute__((ext_vector_type(8)))  float  v8f;

__device__ __forceinline__ v8f wmma_bf16(v16bf a, v16bf b, v8f c) {
  // (neg_a, A, neg_b, B, c_mod, C, reuse_a, reuse_b)
  return __builtin_amdgcn_wmma_f32_16x16x32_bf16(false, a, false, b, (short)0, c, false, false);
}

// 16x32 A-fragment from a row-major f32 row pointer (per-lane row m=lane&15).
// Per-lane K pattern: run0 = [ks*32 + hi*8, +8), run1 = [ks*32 + 16 + hi*8, +8).
__device__ __forceinline__ v16bf afrag_f32(const float* row, int ks, int hi8) {
  const int o0 = ks * 32 + hi8;
  const int o1 = ks * 32 + 16 + hi8;
  v16bf r;
#pragma unroll
  for (int i = 0; i < 8; ++i) {
    r[i]     = (__bf16)row[o0 + i];
    r[8 + i] = (__bf16)row[o1 + i];
  }
  return r;
}

// 32x16 B-fragment (or 16x32 A-fragment) from bf16 LDS; rowp points at the
// start of logical column n (contiguous in k). Same per-lane K pattern.
__device__ __forceinline__ v16bf bfrag_lds(const __bf16* rowp, int ks, int hi8) {
  const int o0 = ks * 32 + hi8;
  const int o1 = ks * 32 + 16 + hi8;
  v8bf lo = *(const v8bf*)(rowp + o0);
  v8bf hh = *(const v8bf*)(rowp + o1);
  v16bf r;
#pragma unroll
  for (int i = 0; i < 8; ++i) { r[i] = lo[i]; r[8 + i] = hh[i]; }
  return r;
}

// ---------------- prep kernel: s = strength @ str_w + str_b -> ws ------------
__global__ void prep_s_kernel(const float* __restrict__ strength,
                              const float* __restrict__ str_w,
                              const float* __restrict__ str_b,
                              float* __restrict__ s_out) {
  int n = threadIdx.x;
  if (n < 64) {
    float acc = str_b[n];
    for (int i = 0; i < 512; ++i) acc = fmaf(strength[i], str_w[i * 64 + n], acc);
    s_out[n] = acc;
  }
}

// ---------------- main fused kernel ------------------------------------------
__global__ __launch_bounds__(256, 1)
void fused_attn2d(const float* __restrict__ q,
                  const float* __restrict__ k,
                  const float* __restrict__ pos,
                  const float* __restrict__ q_tbl,
                  const float* __restrict__ k_tbl,
                  const float* __restrict__ v_tbl,
                  const float* __restrict__ pos_w1,
                  const float* __restrict__ pos_b1,
                  const float* __restrict__ pos_w2,
                  const float* __restrict__ pos_b2,
                  const float* __restrict__ attn_w1,
                  const float* __restrict__ attn_b1,
                  const float* __restrict__ attn_w2,
                  const float* __restrict__ attn_b2,
                  const float* __restrict__ out_w,
                  const float* __restrict__ out_b,
                  const int*   __restrict__ mask,
                  const int*   __restrict__ embed_id,
                  const float* __restrict__ s_vec,
                  float*       __restrict__ out) {
  // LDS (~63.6 KB): weight tiles in bf16 + staging buffers.
  __shared__ __bf16 sWk[64 * 64];          // Wk row-major [n][k]
  __shared__ __bf16 sWv[64 * 64];          // Wv row-major [n][k]
  __shared__ __bf16 sWqo[64 * 64];         // Wq first, then out_w^T [n][k]
  __shared__ __bf16 sPw2T[64 * 32];        // pos_w2^T padded: [n][k<32], k>=8 zero
  __shared__ __bf16 sAw1T[16 * 64];        // attn_w1^T padded: [h<16][k<64], h>=8 zero
  __shared__ __bf16 sAw2T[64 * 32];        // attn_w2^T padded: [n][k<32], k>=8 zero
  __shared__ float  sQh[16 * 64];          // qh tile for the 16 (b,n) of this block
  __shared__ __bf16 sStage[8][16 * 64];    // per-wave C->A re-layout staging (attn_pre)
  __shared__ float  sT1[8][16 * 8];        // per-wave relu(hidden) staging
  __shared__ float  sX[16 * 64];           // fused x before out projection
  __shared__ float  sPw1[40];              // pos_w1 (32) + pos_b1 (8), f32

  const int tid  = threadIdx.x;
  const int wave = tid >> 5;
  const int lane = tid & 31;
  const int lo16 = lane & 15;
  const int hi   = lane >> 4;
  const int hi8  = hi * 8;
  const int base_bn = blockIdx.x * 16;            // 16 (b,n) pairs per block
  const long kbase  = (long)blockIdx.x * 128;     // 128 k-rows per block

  const int eid = embed_id[0];
  const float* Wq = q_tbl + (long)eid * 4096;
  const float* Wk = k_tbl + (long)eid * 4096;
  const float* Wv = v_tbl + (long)eid * 4096;

  // ---- fill LDS weight tiles (bf16 conversion on the fly) ----
  for (int i = tid; i < 4096; i += 256) {
    sWk[i]  = (__bf16)Wk[i];
    sWv[i]  = (__bf16)Wv[i];
    sWqo[i] = (__bf16)Wq[i];
  }
  for (int i = tid; i < 64 * 32; i += 256) {
    int n = i >> 5, kk = i & 31;
    sPw2T[i] = (kk < 8) ? (__bf16)pos_w2[kk * 64 + n] : (__bf16)0.f;
    sAw2T[i] = (kk < 8) ? (__bf16)attn_w2[kk * 64 + n] : (__bf16)0.f;
  }
  for (int i = tid; i < 16 * 64; i += 256) {
    int h = i >> 6, kk = i & 63;
    sAw1T[i] = (h < 8) ? (__bf16)attn_w1[kk * 8 + h] : (__bf16)0.f;
  }
  if (tid < 32)      sPw1[tid] = pos_w1[tid];
  else if (tid < 40) sPw1[tid] = pos_b1[tid - 32];
  __syncthreads();

  // ---- wave 0: qh = q @ Wq^T for this block's 16 rows (no +s: cancels in kh-qh)
  if (wave == 0) {
    const float* qrow = q + (long)(base_bn + lo16) * 64;
    v16bf qa0 = afrag_f32(qrow, 0, hi8);
    v16bf qa1 = afrag_f32(qrow, 1, hi8);
#pragma unroll
    for (int nt = 0; nt < 4; ++nt) {
      int n = lo16 + nt * 16;
      v8f c = {};
      c = wmma_bf16(qa0, bfrag_lds(sWqo + n * 64, 0, hi8), c);
      c = wmma_bf16(qa1, bfrag_lds(sWqo + n * 64, 1, hi8), c);
#pragma unroll
      for (int j = 0; j < 8; ++j) sQh[(j + hi8) * 64 + n] = c[j];
    }
  }
  __syncthreads();

  // ---- repurpose sWqo: out_w^T in bf16 (consumed only after next barrier) ----
  for (int i = tid; i < 4096; i += 256) {
    int n = i >> 6, kk = i & 63;
    sWqo[i] = (__bf16)out_w[kk * 64 + n];
  }

  // ---- per-wave main compute: 16 k-rows = 2 (b,n) groups x 8 views ----
  const long krow0 = kbase + wave * 16;
  const float* krow = k + (krow0 + lo16) * 64;
  v16bf a0 = afrag_f32(krow, 0, hi8);
  v16bf a1 = afrag_f32(krow, 1, hi8);

  v8f c_kh[4], c_vh[4], c_p[4];
#pragma unroll
  for (int nt = 0; nt < 4; ++nt) {
    int n = lo16 + nt * 16;
    v8f c = {};
    c = wmma_bf16(a0, bfrag_lds(sWk + n * 64, 0, hi8), c);
    c = wmma_bf16(a1, bfrag_lds(sWk + n * 64, 1, hi8), c);
    c_kh[nt] = c;
    float sv = s_vec[n];                 // vh = k @ Wv^T + s (seed C with s)
    v8f d;
#pragma unroll
    for (int j = 0; j < 8; ++j) d[j] = sv;
    d = wmma_bf16(a0, bfrag_lds(sWv + n * 64, 0, hi8), d);
    d = wmma_bf16(a1, bfrag_lds(sWv + n * 64, 1, hi8), d);
    c_vh[nt] = d;
  }

  // ---- pos MLP: hidden per lane (K=4 -> 8), layer2 via WMMA (K padded to 32)
  float4 pv = *(const float4*)(pos + (krow0 + lo16) * 4);
  float hdn[8];
#pragma unroll
  for (int j = 0; j < 8; ++j) {
    float t = sPw1[32 + j];
    t = fmaf(pv.x, sPw1[0 * 8 + j], t);
    t = fmaf(pv.y, sPw1[1 * 8 + j], t);
    t = fmaf(pv.z, sPw1[2 * 8 + j], t);
    t = fmaf(pv.w, sPw1[3 * 8 + j], t);
    hdn[j] = fmaxf(t, 0.f);
  }
  v16bf ap;
#pragma unroll
  for (int j = 0; j < 8; ++j) {
    ap[j]     = (hi == 0) ? (__bf16)hdn[j] : (__bf16)0.f;  // k=0..7 real, rest zero
    ap[8 + j] = (__bf16)0.f;
  }
#pragma unroll
  for (int nt = 0; nt < 4; ++nt) {
    int n = lo16 + nt * 16;
    float b2 = pos_b2[n];
    v8f c;
#pragma unroll
    for (int j = 0; j < 8; ++j) c[j] = b2;
    c_p[nt] = wmma_bf16(ap, bfrag_lds(sPw2T + n * 32, 0, hi8), c);
  }

  // ---- attn_pre = kh - qh + p ; stage to LDS for C->A re-layout ----
  __bf16* stg = sStage[wave];
#pragma unroll
  for (int nt = 0; nt < 4; ++nt) {
    int n = lo16 + nt * 16;
    float qh = sQh[(2 * wave + hi) * 64 + n];
#pragma unroll
    for (int j = 0; j < 8; ++j)
      stg[(j + hi8) * 64 + n] = (__bf16)(c_kh[nt][j] - qh + c_p[nt][j]);
  }
  asm volatile("s_wait_dscnt 0x0" ::: "memory");   // wave-local LDS RAW

  // ---- attn MLP layer 1: [16x64] @ attn_w1[64x8] (cols 8..15 garbage/zero) ----
  v16bf sa0 = bfrag_lds(stg + lo16 * 64, 0, hi8);
  v16bf sa1 = bfrag_lds(stg + lo16 * 64, 1, hi8);
  float b1 = attn_b1[lo16 & 7];
  v8f t1;
#pragma unroll
  for (int j = 0; j < 8; ++j) t1[j] = b1;
  t1 = wmma_bf16(sa0, bfrag_lds(sAw1T + lo16 * 64, 0, hi8), t1);
  t1 = wmma_bf16(sa1, bfrag_lds(sAw1T + lo16 * 64, 1, hi8), t1);
  float* tst = sT1[wave];
  if (lo16 < 8) {
#pragma unroll
    for (int j = 0; j < 8; ++j) tst[(j + hi8) * 8 + lo16] = fmaxf(t1[j], 0.f);
  }
  asm volatile("s_wait_dscnt 0x0" ::: "memory");

  // ---- attn MLP layer 2 A-fragment (K padded: k>=8 zero) ----
  v16bf at;
#pragma unroll
  for (int j = 0; j < 8; ++j) {
    float tv = tst[lo16 * 8 + j];
    at[j]     = (hi == 0) ? (__bf16)tv : (__bf16)0.f;
    at[8 + j] = (__bf16)0.f;
  }

  int m8[8];
#pragma unroll
  for (int j = 0; j < 8; ++j) m8[j] = mask[krow0 + j + hi8];

  // ---- layer2 WMMA + mask + softmax over V (pure per-lane, over 8 C VGPRs) ----
#pragma unroll
  for (int nt = 0; nt < 4; ++nt) {
    int n = lo16 + nt * 16;
    float b2 = attn_b2[n];
    v8f c;
#pragma unroll
    for (int j = 0; j < 8; ++j) c[j] = b2;
    c = wmma_bf16(at, bfrag_lds(sAw2T + n * 32, 0, hi8), c);

    float sc[8];
    float mx = -3.4e38f;
#pragma unroll
    for (int j = 0; j < 8; ++j) {
      sc[j] = (m8[j] == 0) ? -1e9f : c[j];
      mx = fmaxf(mx, sc[j]);
    }
    float ssum = 0.f;
#pragma unroll
    for (int j = 0; j < 8; ++j) { sc[j] = __expf(sc[j] - mx); ssum += sc[j]; }
    float inv = 1.f / ssum;
    float x = 0.f;
#pragma unroll
    for (int j = 0; j < 8; ++j)
      x = fmaf(c_vh[nt][j] + c_p[nt][j], sc[j] * inv, x);
    sX[(2 * wave + hi) * 64 + n] = x;     // row = local (b,n), col = channel
  }
  __syncthreads();   // sX complete; sWqo now holds out_w^T

  // ---- out projection: waves 0..3 each own one 16-col tile ----
  if (wave < 4) {
    int n = lo16 + wave * 16;
    v16bf xa0 = afrag_f32(sX + lo16 * 64, 0, hi8);
    v16bf xa1 = afrag_f32(sX + lo16 * 64, 1, hi8);
    float ob = out_b[n];
    v8f c;
#pragma unroll
    for (int j = 0; j < 8; ++j) c[j] = ob;
    c = wmma_bf16(xa0, bfrag_lds(sWqo + n * 64, 0, hi8), c);
    c = wmma_bf16(xa1, bfrag_lds(sWqo + n * 64, 1, hi8), c);
#pragma unroll
    for (int j = 0; j < 8; ++j)
      out[(long)(base_bn + j + hi8) * 64 + n] = c[j];
  }
}

extern "C" void kernel_launch(void* const* d_in, const int* in_sizes, int n_in,
                              void* d_out, int out_size, void* d_ws, size_t ws_size,
                              hipStream_t stream) {
  (void)in_sizes; (void)n_in; (void)out_size; (void)ws_size;
  const float* q        = (const float*)d_in[0];
  const float* k        = (const float*)d_in[1];
  const float* pos      = (const float*)d_in[2];
  const float* strength = (const float*)d_in[3];
  const float* q_tbl    = (const float*)d_in[4];
  const float* k_tbl    = (const float*)d_in[5];
  const float* v_tbl    = (const float*)d_in[6];
  const float* pos_w1   = (const float*)d_in[7];
  const float* pos_b1   = (const float*)d_in[8];
  const float* pos_w2   = (const float*)d_in[9];
  const float* pos_b2   = (const float*)d_in[10];
  const float* attn_w1  = (const float*)d_in[11];
  const float* attn_b1  = (const float*)d_in[12];
  const float* attn_w2  = (const float*)d_in[13];
  const float* attn_b2  = (const float*)d_in[14];
  const float* out_w    = (const float*)d_in[15];
  const float* out_b    = (const float*)d_in[16];
  const float* str_w    = (const float*)d_in[17];
  const float* str_b    = (const float*)d_in[18];
  const int*   mask     = (const int*)d_in[19];
  const int*   embed_id = (const int*)d_in[20];
  float* out   = (float*)d_out;
  float* s_vec = (float*)d_ws;   // 64 floats

  prep_s_kernel<<<1, 64, 0, stream>>>(strength, str_w, str_b, s_vec);
  fused_attn2d<<<BN_TOTAL / 16, 256, 0, stream>>>(
      q, k, pos, q_tbl, k_tbl, v_tbl,
      pos_w1, pos_b1, pos_w2, pos_b2,
      attn_w1, attn_b1, attn_w2, attn_b2,
      out_w, out_b, mask, embed_id, s_vec, out);
}
